// Warp_49039936586102
// MI455X (gfx1250) — matrix operand order repaired and checked by reading, not compiled, not run on hardware
//
#include <hip/hip_runtime.h>

// ---------- WMMA types (gfx1250, wave32) ----------
typedef __attribute__((ext_vector_type(16))) __bf16 v16bf;
typedef __attribute__((ext_vector_type(8)))  float  v8f;

union FragBF { v16bf v; unsigned short u[16]; uint4 q[2]; };

__device__ __forceinline__ unsigned short f2bf(float f) {
    union { float f; unsigned int u; } c; c.f = f;
    unsigned int u = c.u;
    unsigned int r = u + 0x7FFFu + ((u >> 16) & 1u);   // round-to-nearest-even
    return (unsigned short)(r >> 16);
}

#define HW_SHIFT 14          // H*W = 16384
#define W_SHIFT  7           // W = 128
#define P9       (8*9*128*128)
#define CPAD     72          // c-dim padding: 144B rows, 16B aligned

// =====================================================================
// Weight prep: fp32 OIHW -> bf16 [col][576], K = tap*64 + c (tap-major)
// rows 0..31  : w_off (27 valid, pad 32)
// rows 32..95 : w_def (64)
// rows 96..111: w1    (9 valid, pad 16)
// =====================================================================
__global__ void k_prep(const float* __restrict__ w_off,
                       const float* __restrict__ w_def,
                       const float* __restrict__ w1,
                       unsigned short* __restrict__ wb) {
    int t = blockIdx.x * 256 + threadIdx.x;
    if (t >= 112 * 576) return;
    int row = t / 576, K = t - row * 576;
    int tap = K >> 6, c = K & 63;
    int src = c * 9 + tap;                 // OIHW flat index within one out-channel
    float v = 0.f;
    if (row < 32)       { if (row < 27)      v = w_off[row * 576 + src]; }
    else if (row < 96)  {                    v = w_def[(row - 32) * 576 + src]; }
    else                { if (row - 96 < 9)  v = w1[(row - 96) * 576 + src]; }
    wb[t] = f2bf(v);
}

// ---------- shared fragment helpers ----------
__device__ __forceinline__ void load_B2(FragBF& B, const unsigned short* __restrict__ wb,
                                        int col, int j, int half) {
    const unsigned short* bp = wb + col * 576 + j * 32 + (half << 3);
    B.q[0] = *(const uint4*)(bp);
    B.q[1] = *(const uint4*)(bp + 16);
}

// A fragment from LDS patch [ky][xx][CPAD] bf16; K = tap*64 + c
__device__ __forceinline__ void load_A_patch(FragBF& A, const unsigned short* patch,
                                             int j, int half, int px) {
    int K0  = j * 32 + (half << 3);
    int tap = K0 >> 6;
    int c0  = K0 & 63;
    int ky  = tap / 3, kx = tap - ky * 3;
    const unsigned short* p = patch + (ky * 132 + px + kx) * CPAD + c0;
    A.q[0] = *(const uint4*)(p);           // K = K0   .. K0+7   (c contiguous)
    A.q[1] = *(const uint4*)(p + 16);      // K = K0+16.. K0+23
}

// =====================================================================
// conv3x3 -> 27 channels (implicit GEMM via WMMA bf16)
// one block per (n,y) row; 8 waves x 16 pixels; 2 N-tiles (32 cols)
// =====================================================================
__global__ __launch_bounds__(256) void k_conv27(const float* __restrict__ x,
                                                const unsigned short* __restrict__ wb,
                                                float* __restrict__ off_ws,
                                                float* __restrict__ filt_out) {
    __shared__ __align__(16) unsigned short patch[3 * 132 * CPAD]; // [ky][xx][c]
    int blk = blockIdx.x;
    int n = blk >> 7, y = blk & 127;
    int tid = threadIdx.x;

    __builtin_prefetch(x + ((size_t)(n * 64) << HW_SHIFT) + (y << W_SHIFT), 0, 0);

    for (int t = tid; t < 64 * 3 * 130; t += 256) {
        int c = t / 390; int rem = t - c * 390;
        int r = rem / 130; int xx = rem - r * 130;
        int gy = y - 1 + r, gx = xx - 1;
        float v = 0.f;
        if ((unsigned)gy < 128u && (unsigned)gx < 128u)
            v = x[((n * 64 + c) << HW_SHIFT) + (gy << W_SHIFT) + gx];
        patch[(r * 132 + xx) * CPAD + c] = f2bf(v);
    }
    __syncthreads();

    int wave = tid >> 5, lane = tid & 31;
    int half = lane >> 4, lrow = lane & 15;
    int x0 = wave << 4;
    int px = x0 + lrow;

    v8f acc0 = {}; v8f acc1 = {};
    for (int j = 0; j < 18; ++j) {
        FragBF A, B0, B1;
        load_A_patch(A, patch, j, half, px);
        load_B2(B0, wb, lrow,      j, half);
        load_B2(B1, wb, 16 + lrow, j, half);
        acc0 = __builtin_amdgcn_wmma_f32_16x16x32_bf16(false, A.v, false, B0.v,
                                                       (short)0, acc0, false, false);
        acc1 = __builtin_amdgcn_wmma_f32_16x16x32_bf16(false, A.v, false, B1.v,
                                                       (short)0, acc1, false, false);
    }

    int col0 = lrow, col1 = 16 + lrow;
    int rowbase = (y << W_SHIFT) + x0 + (half << 3);
    // cols 0..15 are always offsets
    float* p0 = off_ws + ((n * 18 + col0) << HW_SHIFT) + rowbase;
    #pragma unroll
    for (int d = 0; d < 8; ++d) p0[d] = acc0[d];
    // cols 16..31: 16,17 -> offsets; 18..26 -> filt; 27..31 dropped
    if (col1 < 18) {
        float* p1 = off_ws + ((n * 18 + col1) << HW_SHIFT) + rowbase;
        #pragma unroll
        for (int d = 0; d < 8; ++d) p1[d] = acc1[d];
    } else if (col1 < 27) {
        float* p1 = filt_out + ((n * 9 + (col1 - 18)) << HW_SHIFT) + rowbase;
        #pragma unroll
        for (int d = 0; d < 8; ++d) p1[d] = acc1[d];
    }
}

// =====================================================================
// deformable conv: bilinear gathers feeding bf16 WMMA, K = tap*64+c,
// 64 output channels (4 N-tiles). xr = x + out.
// =====================================================================
__global__ __launch_bounds__(256) void k_deform(const float* __restrict__ x,
                                                const float* __restrict__ off_ws,
                                                const unsigned short* __restrict__ wb,
                                                float* __restrict__ xr) {
    __shared__ __align__(16) int   tIdx[8 * 16 * 9 * 4];
    __shared__ __align__(16) float tW  [8 * 16 * 9 * 4];
    int blk = blockIdx.x;
    int n = blk >> 7, y = blk & 127;
    int tid = threadIdx.x, wave = tid >> 5, lane = tid & 31;
    int half = lane >> 4, lrow = lane & 15;
    int x0 = wave << 4;
    int px = x0 + lrow;

    // tap precompute: lanes 0..15 take taps 0..4, lanes 16..31 take taps 5..8
    {
        int kbeg = half ? 5 : 0, kend = half ? 9 : 5;
        for (int k = kbeg; k < kend; ++k) {
            float offy = off_ws[((n * 18 + 2 * k)     << HW_SHIFT) + (y << W_SHIFT) + px];
            float offx = off_ws[((n * 18 + 2 * k + 1) << HW_SHIFT) + (y << W_SHIFT) + px];
            float py  = (float)y  + (float)(k / 3 - 1) + offy;
            float pxx = (float)px + (float)(k % 3 - 1) + offx;
            float y0f = floorf(py),  x0f = floorf(pxx);
            float wy1 = py - y0f,    wx1 = pxx - x0f;
            int yi = (int)y0f, xi = (int)x0f;
            int t = ((wave * 16 + lrow) * 9 + k) * 4;
            float wc[4] = { (1.f - wy1) * (1.f - wx1), (1.f - wy1) * wx1,
                            wy1 * (1.f - wx1),         wy1 * wx1 };
            int yc[4] = { yi, yi, yi + 1, yi + 1 };
            int xc[4] = { xi, xi + 1, xi, xi + 1 };
            #pragma unroll
            for (int q = 0; q < 4; ++q) {
                bool valid = ((unsigned)yc[q] < 128u) && ((unsigned)xc[q] < 128u);
                int yy = min(max(yc[q], 0), 127);
                int xx = min(max(xc[q], 0), 127);
                tIdx[t + q] = (yy << W_SHIFT) + xx;
                tW  [t + q] = valid ? wc[q] : 0.f;
            }
        }
    }
    __syncthreads();

    v8f acc[4] = { v8f{}, v8f{}, v8f{}, v8f{} };
    for (int j = 0; j < 18; ++j) {
        FragBF A;
        int K0  = j * 32 + (half << 3);
        int tap = K0 >> 6;                  // constant within this chunk
        int t   = ((wave * 16 + lrow) * 9 + tap) * 4;
        int4   id = *(const int4*)  &tIdx[t];
        float4 w  = *(const float4*)&tW[t];
        #pragma unroll
        for (int g = 0; g < 2; ++g) {       // two 8-channel runs: c0 and c0+16
            int c0 = (K0 & 63) + (g << 4);
            const float* xp = x + ((n * 64 + c0) << HW_SHIFT);
            #pragma unroll
            for (int e = 0; e < 8; ++e) {
                float v = w.x * xp[id.x] + w.y * xp[id.y]
                        + w.z * xp[id.z] + w.w * xp[id.w];
                A.u[g * 8 + e] = f2bf(v);
                xp += (1 << HW_SHIFT);
            }
        }
        #pragma unroll
        for (int tc = 0; tc < 4; ++tc) {
            FragBF B;
            load_B2(B, wb, tc * 16 + lrow, j, half);
            acc[tc] = __builtin_amdgcn_wmma_f32_16x16x32_bf16(false, A.v, false, B.v,
                                                              (short)0, acc[tc], false, false);
        }
    }

    int rowbase = (y << W_SHIFT) + x0 + (half << 3);
    #pragma unroll
    for (int tc = 0; tc < 4; ++tc) {
        int col = tc * 16 + lrow;
        const float* xi = x  + ((n * 64 + col) << HW_SHIFT) + rowbase;
        float*       xo = xr + ((n * 64 + col) << HW_SHIFT) + rowbase;
        #pragma unroll
        for (int d = 0; d < 8; ++d) xo[d] = acc[tc][d] + xi[d];
    }
}

// =====================================================================
// conv3x3(xr) -> 9 channels + BN + ReLU  (1 N-tile, cols 0..8 valid)
// =====================================================================
__global__ __launch_bounds__(256) void k_conv9(const float* __restrict__ xr,
                                               const unsigned short* __restrict__ wb,
                                               const float* __restrict__ gamma,
                                               const float* __restrict__ beta,
                                               const float* __restrict__ mean,
                                               const float* __restrict__ var,
                                               float* __restrict__ h_out) {
    __shared__ __align__(16) unsigned short patch[3 * 132 * CPAD];
    int blk = blockIdx.x;
    int n = blk >> 7, y = blk & 127;
    int tid = threadIdx.x;

    __builtin_prefetch(xr + ((size_t)(n * 64) << HW_SHIFT) + (y << W_SHIFT), 0, 0);

    for (int t = tid; t < 64 * 3 * 130; t += 256) {
        int c = t / 390; int rem = t - c * 390;
        int r = rem / 130; int xx = rem - r * 130;
        int gy = y - 1 + r, gx = xx - 1;
        float v = 0.f;
        if ((unsigned)gy < 128u && (unsigned)gx < 128u)
            v = xr[((n * 64 + c) << HW_SHIFT) + (gy << W_SHIFT) + gx];
        patch[(r * 132 + xx) * CPAD + c] = f2bf(v);
    }
    __syncthreads();

    int wave = tid >> 5, lane = tid & 31;
    int half = lane >> 4, lrow = lane & 15;
    int x0 = wave << 4;
    int px = x0 + lrow;

    v8f acc = {};
    for (int j = 0; j < 18; ++j) {
        FragBF A, B;
        load_A_patch(A, patch, j, half, px);
        load_B2(B, wb, lrow, j, half);
        acc = __builtin_amdgcn_wmma_f32_16x16x32_bf16(false, A.v, false, B.v,
                                                      (short)0, acc, false, false);
    }

    int col = lrow;
    if (col < 9) {
        float inv = rsqrtf(var[col] + 1e-5f);
        float sc  = gamma[col] * inv;
        float bs  = beta[col] - mean[col] * sc;
        float* po = h_out + ((n * 9 + col) << HW_SHIFT) + (y << W_SHIFT) + x0 + (half << 3);
        #pragma unroll
        for (int d = 0; d < 8; ++d)
            po[d] = fmaxf(acc[d] * sc + bs, 0.f);
    }
}

// =====================================================================
extern "C" void kernel_launch(void* const* d_in, const int* in_sizes, int n_in,
                              void* d_out, int out_size, void* d_ws, size_t ws_size,
                              hipStream_t stream) {
    (void)in_sizes; (void)n_in; (void)out_size; (void)ws_size;
    const float* x     = (const float*)d_in[0];
    const float* w_off = (const float*)d_in[1];
    const float* w_def = (const float*)d_in[2];
    const float* w1    = (const float*)d_in[3];
    const float* gamma = (const float*)d_in[4];
    const float* beta  = (const float*)d_in[5];
    const float* mean  = (const float*)d_in[6];
    const float* var   = (const float*)d_in[7];

    float* out      = (float*)d_out;
    float* h_out    = out;          // 8*9*128*128
    float* filt_out = out + P9;     // 8*9*128*128

    char* ws = (char*)d_ws;
    const size_t OFF_BYTES = (size_t)8 * 18 * 128 * 128 * 4;   // 9,437,184
    const size_t XR_BYTES  = (size_t)8 * 64 * 128 * 128 * 4;   // 33,554,432
    float*          off_ws = (float*)ws;
    float*          xr     = (float*)(ws + OFF_BYTES);
    unsigned short* wb     = (unsigned short*)(ws + OFF_BYTES + XR_BYTES);

    k_prep  <<<(112 * 576 + 255) / 256, 256, 0, stream>>>(w_off, w_def, w1, wb);
    k_conv27<<<1024, 256, 0, stream>>>(x, wb, off_ws, filt_out);
    k_deform<<<1024, 256, 0, stream>>>(x, off_ws, wb + 32 * 576, xr);
    k_conv9 <<<1024, 256, 0, stream>>>(xr, wb + 96 * 576, gamma, beta, mean, var, h_out);
}
